// HOReIDGraphMatchingNet_54065048322077
// MI455X (gfx1250) — compile-verified, matrix operand order checked by reading, not run on hardware
//
#include <hip/hip_runtime.h>
#include <hip/hip_bf16.h>

typedef __attribute__((ext_vector_type(16))) __bf16 v16bf;
typedef __attribute__((ext_vector_type(8)))  float  v8f;
typedef __attribute__((ext_vector_type(4)))  int    v4i;

constexpr int kB    = 2048;
constexpr int kN    = 14;
constexpr int kInF  = 2048;
constexpr int kOutF = 1024;
constexpr float kAlpha = 200.0f;
constexpr float kEps   = 1e-10f;

#if defined(__gfx1250__) && __has_builtin(__builtin_amdgcn_global_load_async_to_lds_b128) && __has_builtin(__builtin_amdgcn_s_wait_asynccnt)
#define USE_ASYNC_LDS 1
#else
#define USE_ASYNC_LDS 0
#endif

// K index held by fragment element e for lane-half lh (16-bit A/B operand,
// 16x16x32 WMMA): VGPR v holds K pairs {0..7,16..23}+8*lh per ISA layout table.
__device__ __forceinline__ int kfrag(int e, int lh) {
    int v = e >> 1, p = e & 1;
    int k = (v < 4) ? (2 * v + p) : (16 + 2 * (v - 4) + p);
    return k + 8 * lh;
}

// ---------------------------------------------------------------------------
// Kernel 0: convert f32 weights -> bf16, pre-swizzled into WMMA B-fragment
// order: [ktile][coltile][lane][elem] so a lane's 16 values are contiguous.
// ---------------------------------------------------------------------------
__global__ __launch_bounds__(256) void pack_weights(
    const float* __restrict__ aw, const float* __restrict__ uw,
    const float* __restrict__ aff,
    __bf16* __restrict__ wApk, __bf16* __restrict__ wUpk,
    __bf16* __restrict__ affpk)
{
    int stride = gridDim.x * 256;
    for (int idx = blockIdx.x * 256 + threadIdx.x; idx < kInF * kOutF; idx += stride) {
        int e = idx & 15, lane = (idx >> 4) & 31, ct = (idx >> 9) & 63, kt = idx >> 15;
        int K = kt * 32 + kfrag(e, lane >> 4);
        int N = ct * 16 + (lane & 15);
        size_t src = (size_t)K * kOutF + N;
        wApk[idx] = (__bf16)aw[src];
        wUpk[idx] = (__bf16)uw[src];
    }
    for (int idx = blockIdx.x * 256 + threadIdx.x; idx < kOutF * kOutF; idx += stride) {
        int e = idx & 15, lane = (idx >> 4) & 31, ct = (idx >> 9) & 63, kt = idx >> 15;
        int K = kt * 32 + kfrag(e, lane >> 4);
        int N = ct * 16 + (lane & 15);
        affpk[idx] = (__bf16)aff[(size_t)K * kOutF + N];
    }
}

// ---------------------------------------------------------------------------
// Kernel 1: Gconv = colnorm(adj) @ relu(x@Wa+ba) + relu(x@Wu+bu)
// One WG per (batch, 256-col span); 4 waves x 4 sub-tiles each -> 8 WMMA
// chains per wave per K-chunk; the staged A fragment is reused 8x.
// e1 is emitted pre-packed in A-fragment order (for the affinity GEMM),
// e2 is emitted flat bf16.
// ---------------------------------------------------------------------------
__global__ __launch_bounds__(128) void gconv_wmma(
    const float* __restrict__ feat,   // [B,14,2048]
    const float* __restrict__ adj,    // [B,14,14]
    const __bf16* __restrict__ wApk, const float* __restrict__ bA,
    const __bf16* __restrict__ wUpk, const float* __restrict__ bU,
    __bf16* __restrict__ e_pack,      // [B][32 kt][32 lane][16 e]
    __bf16* __restrict__ e_flat,      // [B,14,1024]
    int packA)
{
    __shared__ float Anorm[kN][kN];
    __shared__ float nf[kN];
    __shared__ alignas(32) __bf16 featP[32][16];   // A fragment, packed per lane
    __shared__ float axL[4][16][16];
    __shared__ float uxL[4][16][16];

    int b    = blockIdx.y;
    int tid  = threadIdx.x;
    int wave = tid >> 5;
    int lane = tid & 31;
    int lh   = lane >> 4;
    int lr   = lane & 15;

    // column-L1 normalization of adjacency (sum over rows -> per-column)
    const float* adjb = adj + (size_t)b * kN * kN;
    if (tid < kN) {
        float s = 0.f;
        for (int i = 0; i < kN; ++i) s += adjb[i * kN + tid];
        nf[tid] = 1.0f / fmaxf(s, 1e-12f);
    }
    __syncthreads();
    if (tid < kN * kN) {
        int i = tid / kN, j = tid % kN;
        Anorm[i][j] = adjb[i * kN + j] * nf[j];
    }

    const float* fb = feat + (size_t)b * kN * kInF;
    v8f accA[4] = {};
    v8f accU[4] = {};

    for (int k0 = 0; k0 < kInF; k0 += 32) {
        __syncthreads();   // previous fragment reads complete
        // stage feat chunk directly into A-fragment layout (coalesced reads),
        // prefetch the next chunk behind the WMMA work
        for (int idx = tid; idx < 16 * 32; idx += 128) {
            int r = idx >> 5, c = idx & 31;
            float v = 0.f;
            if (r < kN) {
                v = fb[(size_t)r * kInF + k0 + c];
                if (k0 + 32 < kInF)
                    __builtin_prefetch(&fb[(size_t)r * kInF + k0 + 32 + c], 0, 3);
            }
            int half = (c >> 3) & 1;
            int e    = (c & 7) + ((c & 16) >> 1);
            featP[r + 16 * half][e] = (__bf16)v;
        }
        __syncthreads();

        v16bf afrag = *(const v16bf*)(&featP[lane][0]);
        #pragma unroll
        for (int j = 0; j < 4; ++j) {
            int ct = blockIdx.x * 16 + j * 4 + wave;
            size_t bOff = (((size_t)(k0 >> 5) * 64 + ct) * 32 + lane) * 16;
            v16bf bAfrag = *(const v16bf*)(wApk + bOff);
            v16bf bUfrag = *(const v16bf*)(wUpk + bOff);
            accA[j] = __builtin_amdgcn_wmma_f32_16x16x32_bf16(false, afrag, false, bAfrag,
                                                              (short)0, accA[j], false, false);
            accU[j] = __builtin_amdgcn_wmma_f32_16x16x32_bf16(false, afrag, false, bUfrag,
                                                              (short)0, accU[j], false, false);
        }
    }

    // epilogue per sub-tile group: bias + relu, 14x14 graph mix, emit e
    for (int j = 0; j < 4; ++j) {
        int ct   = blockIdx.x * 16 + j * 4 + wave;
        int colW = ct * 16;
        float biasA = bA[colW + lr];
        float biasU = bU[colW + lr];
        __syncthreads();   // previous sub-tile's combine reads done
        #pragma unroll
        for (int v = 0; v < 8; ++v) {
            int m = v + 8 * lh;
            axL[wave][m][lr] = fmaxf(accA[j][v] + biasA, 0.f);
            uxL[wave][m][lr] = fmaxf(accU[j][v] + biasU, 0.f);
        }
        __syncthreads();

        int col0 = blockIdx.x * 256 + j * 64;
        for (int idx = tid; idx < kN * 64; idx += 128) {
            int n = idx >> 6;
            int c = idx & 63;
            int w = c >> 4, cl = c & 15;
            float acc = uxL[w][n][cl];
            #pragma unroll
            for (int m = 0; m < kN; ++m) acc += Anorm[n][m] * axL[w][m][cl];
            int d = col0 + c;                      // output column = stage-2 K dim
            if (packA) {
                int kt = d >> 5, cc = d & 31;
                int ln = n + 16 * ((cc >> 3) & 1);
                int ee = (cc & 7) + ((cc & 16) >> 1);
                e_pack[(size_t)b * (32 * 32 * 16) + ((size_t)(kt * 32 + ln)) * 16 + ee] =
                    (__bf16)acc;
            } else {
                e_flat[(size_t)b * kN * kOutF + (size_t)n * kOutF + d] = (__bf16)acc;
            }
        }
    }
}

// ---------------------------------------------------------------------------
// Kernel 2: s_raw[b] = e1 @ aff_A @ e2^T   (per-batch WG, 4 waves)
// e2 tile staged LDS-direct via GLOBAL_LOAD_ASYNC_TO_LDS_B128 when available.
// ---------------------------------------------------------------------------
__global__ __launch_bounds__(128) void affinity_wmma(
    const __bf16* __restrict__ e1p,   // [B][32][32][16] packed A fragments
    const __bf16* __restrict__ e2,    // [B,14,1024]
    const __bf16* __restrict__ affpk, // packed B fragments [32][64][32][16]
    float* __restrict__ sraw)         // [B,14,14]
{
    __shared__ alignas(16) __bf16 e2L[kN * kOutF];
    __shared__ float tL[4][16][16];
    __shared__ float sacc[kN * kN];

    int b = blockIdx.x, tid = threadIdx.x;
    int wave = tid >> 5, lane = tid & 31;
    int lh = lane >> 4, lr = lane & 15;

    const __bf16* e2b = e2 + (size_t)b * kN * kOutF;
#if USE_ASYNC_LDS
    {
        __attribute__((address_space(3))) char* dst =
            (__attribute__((address_space(3))) char*)(char*)&e2L[0];
        __attribute__((address_space(1))) char* src =
            (__attribute__((address_space(1))) char*)(void*)e2b;
        for (int off = tid * 16; off < kN * kOutF * 2; off += 128 * 16) {
            __builtin_amdgcn_global_load_async_to_lds_b128(
                (__attribute__((address_space(1))) v4i*)(src + off),
                (__attribute__((address_space(3))) v4i*)(dst + off), 0, 0);
        }
        __builtin_amdgcn_s_wait_asynccnt(0);
    }
#else
    for (int i = tid; i < kN * kOutF; i += 128) e2L[i] = e2b[i];
#endif
    for (int i = tid; i < kN * kN; i += 128) sacc[i] = 0.f;

    const __bf16* e1b = e1p + (size_t)b * (32 * 32 * 16);

    for (int tile = 0; tile < 16; ++tile) {
        int ct = tile * 4 + wave;                  // aff column tile 0..63
        v8f acc0 = {};
        v8f acc1 = {};
        for (int kt = 0; kt < 32; kt += 2) {
            v16bf a0 = *(const v16bf*)(e1b + ((size_t)(kt * 32 + lane)) * 16);
            v16bf b0 = *(const v16bf*)(affpk + (((size_t)kt * 64 + ct) * 32 + lane) * 16);
            acc0 = __builtin_amdgcn_wmma_f32_16x16x32_bf16(false, a0, false, b0,
                                                           (short)0, acc0, false, false);
            v16bf a1 = *(const v16bf*)(e1b + ((size_t)((kt + 1) * 32 + lane)) * 16);
            v16bf b1 = *(const v16bf*)(affpk + (((size_t)(kt + 1) * 64 + ct) * 32 + lane) * 16);
            acc1 = __builtin_amdgcn_wmma_f32_16x16x32_bf16(false, a1, false, b1,
                                                           (short)0, acc1, false, false);
        }
        v8f acc = acc0 + acc1;

        __syncthreads();   // previous tile's tL consumers done; e2L/sacc visible
        #pragma unroll
        for (int v = 0; v < 8; ++v) tL[wave][v + 8 * lh][lr] = acc[v];
        __syncthreads();

        // s[n,m] += sum_c t[n, tile*64+c] * e2[m, tile*64+c]
        for (int idx = tid; idx < kN * kN; idx += 128) {
            int n = idx / kN, m = idx % kN;
            float a = 0.f;
            #pragma unroll
            for (int c = 0; c < 64; ++c)
                a += tL[c >> 4][n][c & 15] * (float)e2L[m * kOutF + tile * 64 + c];
            sacc[idx] += a;
        }
    }
    __syncthreads();
    for (int i = tid; i < kN * kN; i += 128)
        sraw[(size_t)b * kN * kN + i] = sacc[i];
}

// ---------------------------------------------------------------------------
// Kernel 3: voting softmax (rows) + 20 alternating Sinkhorn normalizations
// ---------------------------------------------------------------------------
__global__ __launch_bounds__(256) void vote_sinkhorn(
    const float* __restrict__ sraw, float* __restrict__ sout)
{
    __shared__ float S[kN][kN];
    __shared__ float dsum[kN];
    int b = blockIdx.x, tid = threadIdx.x;
    if (tid < kN * kN) S[tid / kN][tid % kN] = sraw[(size_t)b * kN * kN + tid];
    __syncthreads();

    if (tid < kN) {
        float mx = -3.4e38f;
        for (int j = 0; j < kN; ++j) mx = fmaxf(mx, kAlpha * S[tid][j]);
        float tmp[kN];
        float sum = 0.f;
        for (int j = 0; j < kN; ++j) { tmp[j] = __expf(kAlpha * S[tid][j] - mx); sum += tmp[j]; }
        float inv = 1.f / sum;
        for (int j = 0; j < kN; ++j) S[tid][j] = tmp[j] * inv;
    }
    __syncthreads();

    for (int it = 0; it < 20; ++it) {
        if ((it & 1) == 0) {
            if (tid < kN) {
                float s = 0.f;
                for (int j = 0; j < kN; ++j) s += S[tid][j];
                dsum[tid] = 1.f / (s + kEps);
            }
            __syncthreads();
            if (tid < kN * kN) S[tid / kN][tid % kN] *= dsum[tid / kN];
        } else {
            if (tid < kN) {
                float s = 0.f;
                for (int i = 0; i < kN; ++i) s += S[i][tid];
                dsum[tid] = 1.f / (s + kEps);
            }
            __syncthreads();
            if (tid < kN * kN) S[tid / kN][tid % kN] *= dsum[tid % kN];
        }
        __syncthreads();
    }
    if (tid < kN * kN) sout[(size_t)b * kN * kN + tid] = S[tid / kN][tid % kN];
}

// ---------------------------------------------------------------------------
// Kernel 4: out1 = feat1 + s@feat2 ; out2 = feat2 + s^T@feat1  (mem-bound)
// ---------------------------------------------------------------------------
__global__ __launch_bounds__(256) void fuse_out(
    const float* __restrict__ feat1, const float* __restrict__ feat2,
    const float* __restrict__ s,
    float* __restrict__ out1, float* __restrict__ out2)
{
    __shared__ float S[kN][kN];
    int b = blockIdx.y, tid = threadIdx.x;
    if (tid < kN * kN) S[tid / kN][tid % kN] = s[(size_t)b * kN * kN + tid];
    __syncthreads();

    int f = blockIdx.x * 256 + tid;
    const float* f1 = feat1 + (size_t)b * kN * kInF + f;
    const float* f2 = feat2 + (size_t)b * kN * kInF + f;
    float a1[kN], a2[kN];
    #pragma unroll
    for (int n = 0; n < kN; ++n) { a1[n] = f1[(size_t)n * kInF]; a2[n] = f2[(size_t)n * kInF]; }

    float* o1 = out1 + (size_t)b * kN * kInF + f;
    float* o2 = out2 + (size_t)b * kN * kInF + f;
    #pragma unroll
    for (int n = 0; n < kN; ++n) {
        float acc = a1[n];
        #pragma unroll
        for (int m = 0; m < kN; ++m) acc += S[n][m] * a2[m];
        o1[(size_t)n * kInF] = acc;
    }
    #pragma unroll
    for (int m = 0; m < kN; ++m) {
        float acc = a2[m];
        #pragma unroll
        for (int n = 0; n < kN; ++n) acc += S[n][m] * a1[n];
        o2[(size_t)m * kInF] = acc;
    }
}

// ---------------------------------------------------------------------------
extern "C" void kernel_launch(void* const* d_in, const int* in_sizes, int n_in,
                              void* d_out, int out_size, void* d_ws, size_t ws_size,
                              hipStream_t stream)
{
    const float* feat1 = (const float*)d_in[0];
    const float* feat2 = (const float*)d_in[1];
    const float* adj   = (const float*)d_in[2];
    const float* a_w   = (const float*)d_in[3];
    const float* a_b   = (const float*)d_in[4];
    const float* u_w   = (const float*)d_in[5];
    const float* u_b   = (const float*)d_in[6];
    const float* aff   = (const float*)d_in[7];

    char* ws = (char*)d_ws;
    const size_t MB = 1024 * 1024;
    __bf16* wApk  = (__bf16*)(ws);               //  4 MB packed bf16 a_fc_w
    __bf16* wUpk  = (__bf16*)(ws + 4 * MB);      //  4 MB packed bf16 u_fc_w
    __bf16* affpk = (__bf16*)(ws + 8 * MB);      //  2 MB packed bf16 aff_A
    __bf16* e1p   = (__bf16*)(ws + 10 * MB);     // 64 MB e1 in A-fragment order
    __bf16* e2f   = (__bf16*)(ws + 74 * MB);     // 56 MB e2 flat bf16
    float*  sraw  = (float*)(ws + 130 * MB);     // ~1.6 MB raw affinity

    float* s_out = (float*)d_out;                        // [B,14,14]
    float* out1  = s_out + (size_t)kB * kN * kN;         // [B,14,2048]
    float* out2  = out1 + (size_t)kB * kN * kInF;        // [B,14,2048]

    pack_weights<<<1024, 256, 0, stream>>>(a_w, u_w, aff, wApk, wUpk, affpk);

    dim3 g1(kOutF / 256, kB);
    gconv_wmma<<<g1, 128, 0, stream>>>(feat1, adj, wApk, a_b, wUpk, u_b,
                                       e1p, nullptr, 1);
    gconv_wmma<<<g1, 128, 0, stream>>>(feat2, adj, wApk, a_b, wUpk, u_b,
                                       nullptr, e2f, 0);

    affinity_wmma<<<kB, 128, 0, stream>>>(e1p, e2f, affpk, sraw);
    vote_sinkhorn<<<kB, 256, 0, stream>>>(sraw, s_out);

    dim3 g4(kInF / 256, kB);
    fuse_out<<<g4, 256, 0, stream>>>(feat1, feat2, s_out, out1, out2);
}
